// TSPModel_56195352101336
// MI455X (gfx1250) — matrix operand order repaired and checked by reading, not compiled, not run on hardware
//
#include <hip/hip_runtime.h>

// ---------------------------------------------------------------------------
// TSP model for MI455X (gfx1250, wave32, WMMA).
// GEMMs: bf16 operands, f32 accumulate via v_wmma_f32_16x16x32_bf16,
// 16x32 output strip per wave (A-frag reuse), vector b128 fragment loads,
// software-pipelined K loop (double-buffered fragments) to hide global
// latency in the low-occupancy decoder GEMMs, weights pre-swizzled to
// fragment order. kNN top-k fused per-batch in 320KB WGP LDS.
// ---------------------------------------------------------------------------

typedef __attribute__((ext_vector_type(16))) __bf16          v16bf;
typedef __attribute__((ext_vector_type(16))) unsigned short  v16u;
typedef __attribute__((ext_vector_type(8)))  unsigned short  v8u;
typedef __attribute__((ext_vector_type(8)))  float           v8f;

#define DEV __device__ __forceinline__

DEV unsigned short f32_to_bf16_rne(float f) {
    unsigned int u = __builtin_bit_cast(unsigned int, f);
    u = u + 0x7FFFu + ((u >> 16) & 1u);
    return (unsigned short)(u >> 16);
}

DEV v8f wmma_bf16(v16u a, v16u b, v8f c) {
    return __builtin_amdgcn_wmma_f32_16x16x32_bf16(
        false, __builtin_bit_cast(v16bf, a),
        false, __builtin_bit_cast(v16bf, b),
        (short)0, c, false, false);
}

// ---------------- weight swizzle: KxN f32 -> fragment-ordered bf16 ---------
// out[((tn*(K/32)+kb32)*32 + lane)*16 + e] = bf16(W[(kb32*32 + (lane&16?16:0) + e)*N
//                                                   + tn*16 + (lane&15)])
__global__ void k_swizzleB(const float* __restrict__ W,
                           unsigned short* __restrict__ out, int K, int N) {
    int i = blockIdx.x * 256 + threadIdx.x;
    if (i >= K * N) return;
    int e    = i & 15;
    int lane = (i >> 4) & 31;
    int rest = i >> 9;                       // tn*(K/32) + kb32
    int kt   = K >> 5;
    int tn   = rest / kt, kb32 = rest - tn * kt;
    int k = (kb32 << 5) + ((lane & 16) ? 16 : 0) + e;
    int n = (tn << 4) + (lane & 15);
    out[i] = f32_to_bf16_rne(W[(size_t)k * N + n]);
}

// ---------------- h = coords @ in_W + in_b (f32 + bf16 outputs) ------------
__global__ void k_encode_in(const float* __restrict__ coords,
                            const float* __restrict__ W,
                            const float* __restrict__ bias,
                            float* __restrict__ h,
                            unsigned short* __restrict__ hbf) {
    int r = blockIdx.x, d = threadIdx.x;            // D == 256 == blockDim
    float c0 = coords[2 * r], c1 = coords[2 * r + 1];
    float v = c0 * W[d] + c1 * W[256 + d] + bias[d];
    h[(size_t)r * 256 + d]   = v;
    hbf[(size_t)r * 256 + d] = f32_to_bf16_rne(v);
}

// ---------------- WMMA GEMM: {C,Cbf} = act(A@B + bias) + resid -------------
// A: MxK bf16 row-major (u16). Bsw: fragment-swizzled weight (see k_swizzleB).
// One wave -> 16x32 strip (two 16x16 tiles share the A fragment).
// K loop is a peeled 2-stage pipeline: next fragments load while current
// WMMAs execute (critical for the 1-wave/SIMD decoder GEMMs).
__global__ void k_gemm(const unsigned short* __restrict__ A, int lda,
                       const unsigned short* __restrict__ Bsw,
                       const float* __restrict__ bias,
                       const float* __restrict__ resid, int ldr,
                       float* __restrict__ C,
                       unsigned short* __restrict__ Cbf,
                       int ldc, int M, int N, int K, int act) {
    int wid  = (blockIdx.x * blockDim.x + threadIdx.x) >> 5;
    int lane = threadIdx.x & 31;
    int tn2c = N >> 5;
    int total = (M >> 4) * tn2c;
    if (wid >= total) return;
    int tm = wid / tn2c, t2 = wid - tm * tn2c;
    int tn0 = 2 * t2, tn1 = tn0 + 1;
    int l15 = lane & 15;
    int half8 = (lane & 16) ? 8 : 0;
    int mrow = (tm << 4) + l15;
    const unsigned short* arow = A + (size_t)mrow * lda;
    int kt = K >> 5;
    const unsigned short* bp0 = Bsw + (((size_t)tn0 * kt) << 9) + (lane << 4);
    const unsigned short* bp1 = Bsw + (((size_t)tn1 * kt) << 9) + (lane << 4);

    v8f acc0 = {0.f,0.f,0.f,0.f,0.f,0.f,0.f,0.f};
    v8f acc1 = {0.f,0.f,0.f,0.f,0.f,0.f,0.f,0.f};

    // pipeline stage 0
    v8u  a0 = *(const v8u*)(arow + 2 * half8);      // b128
    v8u  a1 = *(const v8u*)(arow + 8 + 2 * half8);  // b128
    v16u b0 = *(const v16u*)(bp0);                  // 2x b128
    v16u b1 = *(const v16u*)(bp1);

    for (int kb32 = 0; kb32 < kt - 1; ++kb32) {
        int kn = (kb32 + 1) << 5;
        // issue next-stage loads before consuming current fragments
        v8u  na0 = *(const v8u*)(arow + kn + 2 * half8);
        v8u  na1 = *(const v8u*)(arow + kn + 8 + 2 * half8);
        v16u nb0 = *(const v16u*)(bp0 + ((size_t)(kb32 + 1) << 9));
        v16u nb1 = *(const v16u*)(bp1 + ((size_t)(kb32 + 1) << 9));
        v16u au = __builtin_shufflevector(a0, a1,
                    0,1,2,3,4,5,6,7,8,9,10,11,12,13,14,15);
        acc0 = wmma_bf16(au, b0, acc0);
        acc1 = wmma_bf16(au, b1, acc1);
        a0 = na0; a1 = na1; b0 = nb0; b1 = nb1;
    }
    {   // drain
        v16u au = __builtin_shufflevector(a0, a1,
                    0,1,2,3,4,5,6,7,8,9,10,11,12,13,14,15);
        acc0 = wmma_bf16(au, b0, acc0);
        acc1 = wmma_bf16(au, b1, acc1);
    }

    int nc0 = (tn0 << 4) + l15, nc1 = (tn1 << 4) + l15;
#pragma unroll
    for (int r = 0; r < 8; ++r) {
        int row = (tm << 4) + r + half8;
        float v0 = acc0[r], v1 = acc1[r];
        if (bias)     { v0 += bias[nc0];  v1 += bias[nc1]; }
        if (act == 1) { v0 = v0 / (1.f + __expf(-v0));      // silu
                        v1 = v1 / (1.f + __expf(-v1)); }
        if (resid)    { v0 += resid[(size_t)row * ldr + nc0];
                        v1 += resid[(size_t)row * ldr + nc1]; }
        if (C)   { C[(size_t)row * ldc + nc0] = v0;
                   C[(size_t)row * ldc + nc1] = v1; }
        if (Cbf) { Cbf[(size_t)row * ldc + nc0] = f32_to_bf16_rne(v0);
                   Cbf[(size_t)row * ldc + nc1] = f32_to_bf16_rne(v1); }
    }
}

// ---------------- kn = k / max(||k||, 1e-12) (k lives in qkv[...,256:512]) -
__global__ void k_rownorm(const float* __restrict__ qkv,
                          float* __restrict__ kn, int D) {
    int r = blockIdx.x, t = threadIdx.x;
    float v = qkv[(size_t)r * 768 + 256 + t];
    __shared__ float red[256];
    red[t] = v * v; __syncthreads();
    for (int s = 128; s; s >>= 1) { if (t < s) red[t] += red[t + s]; __syncthreads(); }
    float nrm = sqrtf(red[0]);
    kn[(size_t)r * D + t] = v / fmaxf(nrm, 1e-12f);
}

// ---------------- per-batch kNN: sim = kn@kn^T - 1e9*I, top-8 indices ------
// One block per batch; kn[b] (256x256 f32) cached in LDS with stride 257.
__global__ void k_topk(const float* __restrict__ kn, int* __restrict__ idx,
                       int N, int D) {
    extern __shared__ float lds[];                 // N * 257 floats
    int b = blockIdx.x, t = threadIdx.x;
    for (int i = t; i < N * D; i += 256) {
        int n = i >> 8, d = i & 255;
        lds[n * 257 + d] = kn[((size_t)b * N + n) * D + d];
    }
    __syncthreads();
    int wave = t >> 5, lane = t & 31;
    for (int n = wave; n < N; n += 8) {
        float s[8] = {0.f,0.f,0.f,0.f,0.f,0.f,0.f,0.f};
        const float* rn = &lds[n * 257];
        for (int d = 0; d < D; ++d) {
            float a = rn[d];
#pragma unroll
            for (int j = 0; j < 8; ++j) s[j] += a * lds[(lane + j * 32) * 257 + d];
        }
#pragma unroll
        for (int j = 0; j < 8; ++j) if (lane + j * 32 == n) s[j] -= 1e9f;
        for (int kk = 0; kk < 8; ++kk) {
            float bv = s[0]; int bj = 0;
#pragma unroll
            for (int j = 1; j < 8; ++j) if (s[j] > bv) { bv = s[j]; bj = j; }
            int bm = lane + bj * 32;
            for (int off = 16; off; off >>= 1) {
                float ov = __shfl_xor(bv, off);
                int   om = __shfl_xor(bm, off);
                if (ov > bv || (ov == bv && om < bm)) { bv = ov; bm = om; }
            }
            if ((bm & 31) == lane) s[bm >> 5] = -3.402823466e38f;
            if (lane == 0) idx[((size_t)b * N + n) * 8 + kk] = bm;
        }
    }
}

// ---------------- 8-neighbor attention -> bf16 output ----------------------
__global__ void k_attn(const float* __restrict__ qkv, const int* __restrict__ idx,
                       unsigned short* __restrict__ outbf, int N, int D) {
    int r = (blockIdx.x * 256 + threadIdx.x) >> 5;
    int lane = threadIdx.x & 31;
    int b = r >> 8;                                 // N == 256
    const float* qrow = qkv + (size_t)r * 768;
    int ms[8]; float sc[8];
#pragma unroll
    for (int kk = 0; kk < 8; ++kk) {
        int m = idx[(size_t)r * 8 + kk]; ms[kk] = m;
        const float* krow = qkv + ((size_t)(b * N + m)) * 768 + 256;
        float acc = 0.f;
#pragma unroll
        for (int j = 0; j < 8; ++j) acc += qrow[lane + 32 * j] * krow[lane + 32 * j];
        for (int off = 16; off; off >>= 1) acc += __shfl_xor(acc, off);
        sc[kk] = acc * 0.0625f;                     // 1/sqrt(256)
    }
    float mx = sc[0];
#pragma unroll
    for (int kk = 1; kk < 8; ++kk) mx = fmaxf(mx, sc[kk]);
    float sum = 0.f;
#pragma unroll
    for (int kk = 0; kk < 8; ++kk) { sc[kk] = __expf(sc[kk] - mx); sum += sc[kk]; }
    float inv = 1.f / sum;
    float o[8] = {0.f,0.f,0.f,0.f,0.f,0.f,0.f,0.f};
#pragma unroll
    for (int kk = 0; kk < 8; ++kk) {
        const float* vrow = qkv + ((size_t)(b * N + ms[kk])) * 768 + 512;
        float w = sc[kk] * inv;
#pragma unroll
        for (int j = 0; j < 8; ++j) o[j] += w * vrow[lane + 32 * j];
    }
#pragma unroll
    for (int j = 0; j < 8; ++j)
        outbf[(size_t)r * D + lane + 32 * j] = f32_to_bf16_rne(o[j]);
}

// ---------------- layernorm over D=256 -> bf16 (GEMM A operand) ------------
__global__ void k_layernorm(const float* __restrict__ X,
                            const float* __restrict__ g,
                            const float* __restrict__ bvec,
                            unsigned short* __restrict__ outbf) {
    int r = blockIdx.x, t = threadIdx.x;
    float x = X[(size_t)r * 256 + t];
    __shared__ float red[256];
    red[t] = x; __syncthreads();
    for (int s = 128; s; s >>= 1) { if (t < s) red[t] += red[t + s]; __syncthreads(); }
    float mu = red[0] * (1.f / 256.f); __syncthreads();
    float dx = x - mu;
    red[t] = dx * dx; __syncthreads();
    for (int s = 128; s; s >>= 1) { if (t < s) red[t] += red[t + s]; __syncthreads(); }
    float var = red[0] * (1.f / 256.f);
    float v = dx * rsqrtf(var + 1e-5f) * g[t] + bvec[t];
    outbf[(size_t)r * 256 + t] = f32_to_bf16_rne(v);
}

// ---------------- SSM: x += y; state = tanh(A*state + B*y); bf16(x) --------
__global__ void k_ssm_update(float* __restrict__ x, const float* __restrict__ y,
                             const float* __restrict__ Av, const float* __restrict__ Bv,
                             float* __restrict__ st, unsigned short* __restrict__ xbf,
                             int BD, int D) {
    int i = blockIdx.x * 256 + threadIdx.x;
    if (i >= BD) return;
    int d = i & (D - 1);
    float yy = y[i];
    float nx = x[i] + yy;
    x[i] = nx;
    xbf[i] = f32_to_bf16_rne(nx);
    st[i] = tanhf(Av[d] * st[i] + Bv[d] * yy);
}

// ---------------- logits[b,n] = node_emb[b,n,:] . q[b,:] -------------------
__global__ void k_logits(const float* __restrict__ emb, const float* __restrict__ q,
                         float* __restrict__ logits, int N, int D) {
    int r = (blockIdx.x * 256 + threadIdx.x) >> 5;
    int lane = threadIdx.x & 31;
    int b = r >> 8;                                 // N == 256
    const float* er = emb + (size_t)r * D;
    const float* qb = q + (size_t)b * D;
    float acc = 0.f;
#pragma unroll
    for (int j = 0; j < 8; ++j) acc += er[lane + 32 * j] * qb[lane + 32 * j];
    for (int off = 16; off; off >>= 1) acc += __shfl_xor(acc, off);
    if (lane == 0) logits[r] = acc;
}

// ---------------- decoder select: mask, argmax(+gumbel), logp/ent/tour -----
__global__ void k_select(const float* __restrict__ logits, const float* __restrict__ gum,
                         const float* __restrict__ coords, const float* __restrict__ emb,
                         int* __restrict__ visited, int* __restrict__ current,
                         float* __restrict__ lps, float* __restrict__ ents,
                         float* __restrict__ tl, float* __restrict__ token,
                         int N, int D) {
    int b = blockIdx.x, t = threadIdx.x;
    __shared__ float red[256];
    __shared__ float slogp[256];
    __shared__ float sval[256];
    __shared__ int   sidx[256];
    const float NEG = -3.402823466e38f;
    int vis = visited[b * N + t];
    float lg = vis ? NEG : logits[(size_t)b * N + t];
    red[t] = lg; __syncthreads();
    for (int s = 128; s; s >>= 1) { if (t < s) red[t] = fmaxf(red[t], red[t + s]); __syncthreads(); }
    float mx = red[0]; __syncthreads();
    float ex = __expf(lg - mx);
    red[t] = ex; __syncthreads();
    for (int s = 128; s; s >>= 1) { if (t < s) red[t] += red[t + s]; __syncthreads(); }
    float lZ = mx + __logf(red[0]); __syncthreads();
    float logp = lg - lZ;
    slogp[t] = logp;
    float p = __expf(logp);
    red[t] = (p > 0.f) ? p * logp : 0.f; __syncthreads();
    for (int s = 128; s; s >>= 1) { if (t < s) red[t] += red[t + s]; __syncthreads(); }
    float esum = red[0]; __syncthreads();
    sval[t] = lg + gum[(size_t)b * N + t]; sidx[t] = t; __syncthreads();
    for (int s = 128; s; s >>= 1) {
        if (t < s) {
            if (sval[t + s] > sval[t] ||
                (sval[t + s] == sval[t] && sidx[t + s] < sidx[t])) {
                sval[t] = sval[t + s]; sidx[t] = sidx[t + s];
            }
        }
        __syncthreads();
    }
    int nxt = sidx[0];
    if (t == 0) {
        lps[b]  += slogp[nxt];
        ents[b] -= esum;
        int cur = current[b];
        float dx = coords[(size_t)(b * N + cur) * 2]     - coords[(size_t)(b * N + nxt) * 2];
        float dy = coords[(size_t)(b * N + cur) * 2 + 1] - coords[(size_t)(b * N + nxt) * 2 + 1];
        tl[b] += sqrtf(dx * dx + dy * dy);
        current[b] = nxt;
        visited[b * N + nxt] = 1;
    }
    token[(size_t)b * D + t] = emb[((size_t)b * N + nxt) * D + t];
}

// ---------------- decoder state init ---------------------------------------
__global__ void k_dec_init(const float* __restrict__ emb, float* __restrict__ xbuf,
                           int* __restrict__ visited, int* __restrict__ current,
                           float* __restrict__ lps, float* __restrict__ ents,
                           float* __restrict__ tl, float* __restrict__ states,
                           int N, int D, int BD) {
    int b = blockIdx.x, t = threadIdx.x;
    visited[b * N + t] = (t == 0) ? 1 : 0;
    xbuf[(size_t)b * D + t] = emb[(size_t)b * N * D + t];   // node_emb[:,0,:]
    for (int s = 0; s < 4; ++s) states[(size_t)s * BD + b * D + t] = 0.f;
    if (t == 0) { current[b] = 0; lps[b] = 0.f; ents[b] = 0.f; tl[b] = 0.f; }
}

// ---------------- finalize: close tour, emit outputs -----------------------
__global__ void k_finalize(const float* __restrict__ coords, const int* __restrict__ current,
                           const float* __restrict__ lps, const float* __restrict__ ents,
                           const float* __restrict__ tl, float* __restrict__ out,
                           int B, int N) {
    int b = blockIdx.x * blockDim.x + threadIdx.x;
    if (b >= B) return;
    int cur = current[b];
    float dx = coords[(size_t)(b * N + cur) * 2]     - coords[(size_t)b * N * 2];
    float dy = coords[(size_t)(b * N + cur) * 2 + 1] - coords[(size_t)b * N * 2 + 1];
    out[b]         = lps[b];
    out[B + b]     = ents[b];
    out[2 * B + b] = tl[b] + sqrtf(dx * dx + dy * dy);
}

// ===========================================================================
extern "C" void kernel_launch(void* const* d_in, const int* in_sizes, int n_in,
                              void* d_out, int out_size, void* d_ws, size_t ws_size,
                              hipStream_t stream) {
    (void)in_sizes; (void)n_in; (void)out_size; (void)ws_size;
    const int Bb = 256, Nn = 256, Dd = 256, S = Bb * Nn;

    const float* coords = (const float*)d_in[0];
    const float* gumbel = (const float*)d_in[1];
    const float* in_W   = (const float*)d_in[2];
    const float* in_b   = (const float*)d_in[3];
    const float* qkvW[2] = {(const float*)d_in[4], (const float*)d_in[8]};
    const float* qkvb[2] = {(const float*)d_in[5], (const float*)d_in[9]};
    const float* outW[2] = {(const float*)d_in[6], (const float*)d_in[10]};
    const float* outb[2] = {(const float*)d_in[7], (const float*)d_in[11]};
    const float *ln_g[4], *ln_b[4], *w1[4], *b1[4], *w2[4], *b2[4], *Av[4], *Bv[4];
    for (int i = 0; i < 4; ++i) {
        int base = 12 + 8 * i;
        ln_g[i] = (const float*)d_in[base + 0];
        ln_b[i] = (const float*)d_in[base + 1];
        w1[i]   = (const float*)d_in[base + 2];
        b1[i]   = (const float*)d_in[base + 3];
        w2[i]   = (const float*)d_in[base + 4];
        b2[i]   = (const float*)d_in[base + 5];
        Av[i]   = (const float*)d_in[base + 6];
        Bv[i]   = (const float*)d_in[base + 7];
    }
    const float* queryW = (const float*)d_in[44];
    const float* queryb = (const float*)d_in[45];

    // ---- workspace layout ----
    char* ws = (char*)d_ws;
    const size_t SZ = (size_t)S * Dd * 4;                  // 64 MB
    float* bufA = (float*)(ws);
    float* bufB = (float*)(ws + SZ);
    float* qkv  = (float*)(ws + 2 * SZ);                   // 3*SZ
    float* kn   = (float*)(ws + 5 * SZ);
    unsigned short* bfA = (unsigned short*)(ws + 6 * SZ);            // SZ/2
    unsigned short* bfB = (unsigned short*)(ws + 6 * SZ + SZ / 2);   // SZ/2
    unsigned short* wbf = (unsigned short*)(ws + 7 * SZ);            // 4MB
    char* p = ws + 7 * SZ + ((size_t)4 << 20);
    int*   idx     = (int*)p;   p += (size_t)S * 8 * 4;
    float* logits  = (float*)p; p += (size_t)S * 4;
    float* ybuf    = (float*)p; p += (size_t)Bb * Dd * 4;
    float* xbuf    = (float*)p; p += (size_t)Bb * Dd * 4;
    float* qvec    = (float*)p; p += (size_t)Bb * Dd * 4;
    float* states  = (float*)p; p += (size_t)4 * Bb * Dd * 4;
    int*   visited = (int*)p;   p += (size_t)S * 4;
    int*   current = (int*)p;   p += (size_t)Bb * 4;
    float* lps     = (float*)p; p += (size_t)Bb * 4;
    float* ents    = (float*)p; p += (size_t)Bb * 4;
    float* tl      = (float*)p; p += (size_t)Bb * 4;
    unsigned short* lnbf = (unsigned short*)p; p += (size_t)Bb * Dd * 2;
    unsigned short* t1bf = (unsigned short*)p; p += (size_t)Bb * 2 * Dd * 2;
    unsigned short* xbf  = (unsigned short*)p; p += (size_t)Bb * Dd * 2;

    unsigned short* wqkv[2] = { wbf, wbf + 196608 };
    unsigned short* wout[2] = { wbf + 393216, wbf + 458752 };
    unsigned short* ww1[4], *ww2[4];
    for (int i = 0; i < 4; ++i) { ww1[i] = wbf + 524288  + i * 131072;
                                  ww2[i] = wbf + 1048576 + i * 131072; }
    unsigned short* wqry = wbf + 1572864;

    auto swz = [&](const float* w, unsigned short* dst, int K, int N) {
        k_swizzleB<<<(K * N + 255) / 256, 256, 0, stream>>>(w, dst, K, N);
    };
    // pre-swizzle all weights to fragment-ordered bf16
    for (int l = 0; l < 2; ++l) { swz(qkvW[l], wqkv[l], 256, 768);
                                  swz(outW[l], wout[l], 256, 256); }
    for (int i = 0; i < 4; ++i) { swz(w1[i], ww1[i], 256, 512);
                                  swz(w2[i], ww2[i], 512, 256); }
    swz(queryW, wqry, 256, 256);

    auto gemm = [&](const unsigned short* A, int lda, const unsigned short* Bsw,
                    const float* bias, const float* resid, int ldr,
                    float* C, unsigned short* Cbf, int ldc,
                    int M, int N, int K, int act) {
        int waves = (M / 16) * (N / 32);
        k_gemm<<<(waves + 7) / 8, 256, 0, stream>>>(A, lda, Bsw, bias, resid, ldr,
                                                    C, Cbf, ldc, M, N, K, act);
    };

    // ---- encoder ----
    k_encode_in<<<S, 256, 0, stream>>>(coords, in_W, in_b, bufA, bfA);
    float* hf = bufA; float* hg = bufB;
    for (int l = 0; l < 2; ++l) {
        gemm(bfA, Dd, wqkv[l], qkvb[l], nullptr, 0, qkv, nullptr, 768, S, 768, Dd, 0);
        k_rownorm<<<S, 256, 0, stream>>>(qkv, kn, Dd);
        k_topk<<<Bb, 256, 257 * 256 * 4, stream>>>(kn, idx, Nn, Dd);
        k_attn<<<S / 8, 256, 0, stream>>>(qkv, idx, bfB, Nn, Dd);
        gemm(bfB, Dd, wout[l], outb[l], hf, Dd, hg, bfA, Dd, S, Dd, Dd, 0);
        float* tmp = hf; hf = hg; hg = tmp;
    }
    float* node_emb = hf;

    // ---- decoder scan ----
    k_dec_init<<<Bb, 256, 0, stream>>>(node_emb, xbuf, visited, current, lps, ents, tl,
                                       states, Nn, Dd, Bb * Dd);
    for (int step = 0; step < Nn - 1; ++step) {
        for (int i = 0; i < 4; ++i) {
            k_layernorm<<<Bb, 256, 0, stream>>>(xbuf, ln_g[i], ln_b[i], lnbf);
            gemm(lnbf, Dd, ww1[i], b1[i], nullptr, 0, nullptr, t1bf, 2 * Dd,
                 Bb, 2 * Dd, Dd, 1);
            gemm(t1bf, 2 * Dd, ww2[i], b2[i], nullptr, 0, ybuf, nullptr, Dd,
                 Bb, Dd, 2 * Dd, 0);
            k_ssm_update<<<(Bb * Dd) / 256, 256, 0, stream>>>(
                xbuf, ybuf, Av[i], Bv[i], states + (size_t)i * Bb * Dd, xbf,
                Bb * Dd, Dd);
        }
        gemm(xbf, Dd, wqry, queryb, nullptr, 0, qvec, nullptr, Dd, Bb, Dd, Dd, 0);
        k_logits<<<S / 8, 256, 0, stream>>>(node_emb, qvec, logits, Nn, Dd);
        k_select<<<Bb, 256, 0, stream>>>(logits, gumbel + (size_t)step * S, coords,
                                         node_emb, visited, current, lps, ents, tl,
                                         xbuf, Nn, Dd);
    }
    k_finalize<<<1, 256, 0, stream>>>(coords, current, lps, ents, tl, (float*)d_out, Bb, Nn);
}
